// ChannelWiseTCNBlock_20607253086710
// MI455X (gfx1250) — matrix operand order repaired and checked by reading, not compiled, not run on hardware
//
#include <hip/hip_runtime.h>

typedef __attribute__((ext_vector_type(16))) __bf16 v16bf;
typedef __attribute__((ext_vector_type(8)))  __bf16 v8bf;
typedef __attribute__((ext_vector_type(8)))  float  v8f;

#define B_    32
#define CIN   256
#define COUT  512
#define LEN   2048
#define KT    32     // K step per WMMA
#define MT    128    // o tile per block
#define NT    128    // l tile per block
#define KP    40     // padded LDS K-stride (80B rows -> conflict-free b128 gathers)
#define LT_N  (LEN / NT)          // 16 l-tiles
#define NSLOT (B_ * LT_N)         // 512 partial slots per channel

union V16 { v16bf v; v8bf h[2]; };

// MODE 0: U = depthwise_causal_conv(x), W = pw_w; writes y = W*U + pw_b and
//         per-channel partial sum/sum^2 (deterministic slot per block).
// MODE 1: U = x, W = res_w; epilogue out = relu(relu(y*scale+shift) + W*U + res_b)
template<int MODE>
__global__ __launch_bounds__(256)
void tcn_gemm(const float* __restrict__ x,
              const float* __restrict__ sw,
              const float* __restrict__ W,
              const float* __restrict__ bias,
              const float* __restrict__ yin,
              const float* __restrict__ scale,
              const float* __restrict__ shift,
              float* __restrict__ out,
              float* __restrict__ psum,
              float* __restrict__ psq)
{
    __shared__ __bf16 sA[MT * KP];      // W tile  [o][k]
    __shared__ __bf16 sU[NT * KP];      // U^T tile [l][k]
    __shared__ float  sRed[2][2][MT];   // [sum|sq][wl][o_local]

    const int tid  = threadIdx.x;
    const int lane = tid & 31;
    const int wave = tid >> 5;
    const int wo   = wave & 3;   // o offset = 32*wo
    const int wl   = wave >> 2;  // l offset = 64*wl
    const int l0   = blockIdx.x * NT;
    const int o0   = blockIdx.y * MT;
    const int b    = blockIdx.z;

    float w0 = 0.f, w1 = 0.f, w2 = 0.f;
    if (MODE == 0) { w0 = sw[0]; w1 = sw[1]; w2 = sw[2]; }

    v8f acc[2][4];
    #pragma unroll
    for (int r = 0; r < 2; ++r)
        #pragma unroll
        for (int j = 0; j < 4; ++j) { v8f z = {}; acc[r][j] = z; }

    // staging maps: A: 2 threads/row * 16 elems; U: thread covers 16 l's of one k-row
    const int oA  = tid >> 1;
    const int cA0 = (tid & 1) * 16;
    const int cU  = tid >> 3;          // 0..31
    const int lU0 = (tid & 7) * 16;    // 0..112

    for (int k0 = 0; k0 < CIN; k0 += KT) {
        // prefetch next K-tile sources (global_prefetch_b8)
        if (k0 + KT < CIN) {
            __builtin_prefetch(&W[(size_t)(o0 + oA) * CIN + k0 + KT + cA0], 0, 0);
            __builtin_prefetch(&x[((size_t)b * CIN + k0 + KT + cU) * LEN + l0 + lU0], 0, 0);
        }
        // ---- stage A (weights, f32 -> bf16) ----
        {
            const float* src = &W[(size_t)(o0 + oA) * CIN + k0 + cA0];
            __bf16* dst = &sA[oA * KP + cA0];
            #pragma unroll
            for (int i = 0; i < 16; ++i) dst[i] = (__bf16)src[i];
        }
        // ---- stage U^T (fused dilated causal conv in MODE 0) ----
        {
            const int c = k0 + cU;
            const float* xrow = &x[((size_t)b * CIN + c) * LEN];
            if (MODE == 0) {
                if (l0 == 0) {
                    // only the first l-tile can touch the causal pad
                    #pragma unroll
                    for (int i = 0; i < 16; ++i) {
                        const int ll = lU0 + i;
                        const int gl = ll;
                        float x0  = xrow[gl];
                        float xm2 = (gl >= 2) ? xrow[gl - 2] : 0.f;
                        float xm4 = (gl >= 4) ? xrow[gl - 4] : 0.f;
                        sU[ll * KP + cU] = (__bf16)(w2 * x0 + w1 * xm2 + w0 * xm4);
                    }
                } else {
                    #pragma unroll
                    for (int i = 0; i < 16; ++i) {
                        const int ll = lU0 + i;
                        const int gl = l0 + ll;
                        sU[ll * KP + cU] =
                            (__bf16)(w2 * xrow[gl] + w1 * xrow[gl - 2] + w0 * xrow[gl - 4]);
                    }
                }
            } else {
                #pragma unroll
                for (int i = 0; i < 16; ++i) {
                    const int ll = lU0 + i;
                    sU[ll * KP + cU] = (__bf16)xrow[l0 + ll];
                }
            }
        }
        __syncthreads();

        // ---- fragment gathers (ISA 16-bit A/B VGPR layouts) ----
        const int half = lane >> 4;
        const int mrow = lane & 15;
        V16 aF[2], bF[4];
        #pragma unroll
        for (int r = 0; r < 2; ++r) {
            const __bf16* ap = &sA[(wo * 32 + r * 16 + mrow) * KP + half * 8];
            aF[r].h[0] = *(const v8bf*)(ap);        // K = kb .. kb+7
            aF[r].h[1] = *(const v8bf*)(ap + 16);   // K = kb+16 .. kb+23
        }
        #pragma unroll
        for (int j = 0; j < 4; ++j) {
            const __bf16* bp = &sU[(wl * 64 + j * 16 + mrow) * KP + half * 8];
            bF[j].h[0] = *(const v8bf*)(bp);        // K = kb .. kb+7
            bF[j].h[1] = *(const v8bf*)(bp + 16);   // K = kb+16 .. kb+23
        }

        #pragma unroll
        for (int r = 0; r < 2; ++r)
            #pragma unroll
            for (int j = 0; j < 4; ++j)
                acc[r][j] = __builtin_amdgcn_wmma_f32_16x16x32_bf16(
                    false, aF[r].v, false, bF[j].v, (short)0, acc[r][j], false, false);

        __syncthreads();
    }

    // ---- epilogue (C/D layout: o = base + v + 8*(lane>=16), l = base + lane%16) ----
    if (MODE == 0) {
        #pragma unroll
        for (int r = 0; r < 2; ++r) {
            #pragma unroll
            for (int v = 0; v < 8; ++v) {
                const int ol = wo * 32 + r * 16 + v + 8 * (lane >> 4);
                const int og = o0 + ol;
                const float bo = bias[og];
                float s_ = 0.f, q_ = 0.f;
                #pragma unroll
                for (int j = 0; j < 4; ++j) {
                    const int lg = l0 + wl * 64 + j * 16 + (lane & 15);
                    const size_t idx = ((size_t)b * COUT + og) * LEN + lg;
                    float a = acc[r][j][v] + bo;
                    out[idx] = a;
                    s_ += a; q_ += a * a;
                }
                // reduce across the 16-lane group (l direction)
                #pragma unroll
                for (int m = 1; m < 16; m <<= 1) {
                    s_ += __shfl_xor(s_, m, 32);
                    q_ += __shfl_xor(q_, m, 32);
                }
                if ((lane & 15) == 0) {
                    sRed[0][wl][ol] = s_;
                    sRed[1][wl][ol] = q_;
                }
            }
        }
        __syncthreads();
        if (tid < MT) {
            const int slot = b * LT_N + blockIdx.x;   // deterministic per-block slot
            const size_t og = (size_t)(o0 + tid);
            psum[og * NSLOT + slot] = sRed[0][0][tid] + sRed[0][1][tid];
            psq [og * NSLOT + slot] = sRed[1][0][tid] + sRed[1][1][tid];
        }
    } else {
        #pragma unroll
        for (int r = 0; r < 2; ++r) {
            #pragma unroll
            for (int j = 0; j < 4; ++j) {
                const int lg = l0 + wl * 64 + j * 16 + (lane & 15);
                #pragma unroll
                for (int v = 0; v < 8; ++v) {
                    const int og = o0 + wo * 32 + r * 16 + v + 8 * (lane >> 4);
                    const size_t idx = ((size_t)b * COUT + og) * LEN + lg;
                    float a = acc[r][j][v] + bias[og];
                    float bn = yin[idx] * scale[og] + shift[og];
                    bn = bn > 0.f ? bn : 0.f;
                    float o2 = bn + a;
                    out[idx] = o2 > 0.f ? o2 : 0.f;
                }
            }
        }
    }
}

// Reduce 512 deterministic partials per channel -> scale/shift (tiny: ~2MB read)
__global__ __launch_bounds__(256)
void bn_scale(const float* __restrict__ psum, const float* __restrict__ psq,
              const float* __restrict__ gamma, const float* __restrict__ beta,
              float* __restrict__ scale, float* __restrict__ shift)
{
    __shared__ float ss[256], sq[256];
    const int o = blockIdx.x;
    const size_t base = (size_t)o * NSLOT;
    float s = psum[base + threadIdx.x] + psum[base + threadIdx.x + 256];
    float q = psq [base + threadIdx.x] + psq [base + threadIdx.x + 256];
    ss[threadIdx.x] = s; sq[threadIdx.x] = q;
    __syncthreads();
    for (int st = 128; st > 0; st >>= 1) {
        if (threadIdx.x < st) {
            ss[threadIdx.x] += ss[threadIdx.x + st];
            sq[threadIdx.x] += sq[threadIdx.x + st];
        }
        __syncthreads();
    }
    if (threadIdx.x == 0) {
        const float n = (float)(B_ * LEN);
        float mean = ss[0] / n;
        float var  = sq[0] / n - mean * mean;
        float sc   = gamma[o] * rsqrtf(var + 1e-5f);
        scale[o] = sc;
        shift[o] = beta[o] - mean * sc;
    }
}

extern "C" void kernel_launch(void* const* d_in, const int* in_sizes, int n_in,
                              void* d_out, int out_size, void* d_ws, size_t ws_size,
                              hipStream_t stream)
{
    (void)in_sizes; (void)n_in; (void)out_size; (void)ws_size;
    const float* x     = (const float*)d_in[0];
    const float* sw    = (const float*)d_in[1];
    const float* pw_w  = (const float*)d_in[2];
    const float* pw_b  = (const float*)d_in[3];
    const float* gamma = (const float*)d_in[4];
    const float* beta  = (const float*)d_in[5];
    const float* res_w = (const float*)d_in[6];
    const float* res_b = (const float*)d_in[7];
    float* out = (float*)d_out;

    float* y     = (float*)d_ws;                        // B*COUT*LEN f32
    float* scale = y + (size_t)B_ * COUT * LEN;
    float* shift = scale + COUT;
    float* psum  = shift + COUT;                        // COUT*NSLOT
    float* psq   = psum + (size_t)COUT * NSLOT;         // COUT*NSLOT

    dim3 grid(LT_N, COUT / MT, B_);                     // (16, 4, 32)
    tcn_gemm<0><<<grid, 256, 0, stream>>>(x, sw, pw_w, pw_b,
                                          nullptr, nullptr, nullptr, y, psum, psq);
    bn_scale<<<COUT, 256, 0, stream>>>(psum, psq, gamma, beta, scale, shift);
    tcn_gemm<1><<<grid, 256, 0, stream>>>(x, sw, res_w, res_b,
                                          y, scale, shift, out, nullptr, nullptr);
}
